// Attention_3728031613591
// MI455X (gfx1250) — matrix operand order repaired and checked by reading, not compiled, not run on hardware
//
#include <hip/hip_runtime.h>
#include <hip/hip_bf16.h>
#include <math.h>

// ---------------------------------------------------------------------------
// Single-head attention (no 1/sqrt(d) scale), flash-style, bf16 WMMA compute.
//   EMB=312 (padded to 320), HID=256, B=4, L=4096
// Workspace layout (bytes):
//   [0, 480K)      : Wt  bf16 [3][256][320]   (W transposed, K-padded)
//   [512K, +8MB)   : Q   bf16 [B][L][256]
//   [.. , +8MB)    : K   bf16 [B][L][256]
//   [.. , +8MB)    : Vt  bf16 [B][256][L]
// ---------------------------------------------------------------------------

#define EMB   312
#define KPAD  320
#define HID   256
#define BATCH 4
#define SEQ   4096

typedef __attribute__((ext_vector_type(16))) __bf16 bf16x16;
typedef __attribute__((ext_vector_type(8)))  float  f32x8;

static constexpr size_t WT_OFF = 0;
static constexpr size_t Q_OFF  = 512 * 1024;
static constexpr size_t QSZ    = (size_t)BATCH * SEQ * HID * 2;  // 8 MB
static constexpr size_t K_OFF  = Q_OFF + QSZ;
static constexpr size_t V_OFF  = K_OFF + QSZ;

__device__ __forceinline__ void async_load_b128(unsigned lds_off, const void* gptr) {
  // CDNA5 async LDS-direct load (GV mode, 16 bytes/lane). Tracked by ASYNCcnt.
  asm volatile("global_load_async_to_lds_b128 %0, %1, off"
               :: "v"(lds_off), "v"(gptr) : "memory");
}
__device__ __forceinline__ void wait_async16() {
  // Previous chunk's 16 async loads complete (in-order); current 16 in flight.
  asm volatile("s_wait_asynccnt 16" ::: "memory");
}
__device__ __forceinline__ void wait_async0() {
  asm volatile("s_wait_asynccnt 0" ::: "memory");
}

// ------------------------- kernel 0: weight transpose ----------------------
__global__ void wt_kernel(const float* __restrict__ Wq,
                          const float* __restrict__ Wk,
                          const float* __restrict__ Wv,
                          __bf16* __restrict__ wt) {
  int blk = blockIdx.x;            // 0 .. 3*256-1
  int o   = blk >> 8;
  int col = blk & 255;
  const float* W = (o == 0) ? Wq : (o == 1) ? Wk : Wv;
  int j = threadIdx.x;             // 0 .. 319
  __bf16 v = (__bf16)0.0f;
  if (j < EMB) v = (__bf16)W[(size_t)j * HID + col];
  wt[((size_t)o * HID + col) * KPAD + j] = v;
}

// ------------------------- kernel 1: QKV projection ------------------------
// grid = B*(L/16) blocks, 256 threads (8 waves). Wave w owns cols [32w,32w+32)
// of q/k/v for the block's 16 rows: 6 WMMA accumulators, 10 K-chunks.
__global__ void __launch_bounds__(256)
qkv_proj(const float* __restrict__ emb, const __bf16* __restrict__ wt,
         const float* __restrict__ bq, const float* __restrict__ bk,
         const float* __restrict__ bv,
         __bf16* __restrict__ qout, __bf16* __restrict__ kout,
         __bf16* __restrict__ vtout) {
  int blk  = blockIdx.x;           // 0..1023
  int b    = blk >> 8;
  int row0 = (blk & 255) * 16;
  int tid  = threadIdx.x;

  __shared__ __align__(16) __bf16 sA[16 * KPAD];   // 10 KB, bf16 A tile

  const float* src = emb + ((size_t)b * SEQ + row0) * EMB;
  for (int i = tid; i < 16 * EMB; i += 256) {
    int r = i / EMB, c = i - r * EMB;
    sA[r * KPAD + c] = (__bf16)src[i];
  }
  for (int i = tid; i < 16 * (KPAD - EMB); i += 256) {
    int r = i >> 3, c = EMB + (i & 7);
    sA[r * KPAD + c] = (__bf16)0.0f;
  }
  __syncthreads();

  int wave = tid >> 5, lane = tid & 31;
  int half = lane >> 4;            // 0: lanes 0-15, 1: lanes 16-31
  int l15  = lane & 15;

  f32x8 acc[6];
  #pragma unroll
  for (int t = 0; t < 6; ++t) acc[t] = f32x8{};

  for (int kc = 0; kc < KPAD / 32; ++kc) {
    // A operand: lane l<16 -> row l, K={b..b+15}, b = 32kc (+16 for hi half)
    bf16x16 a;
    const __bf16* a0 = &sA[l15 * KPAD + kc * 32 + half * 16];
    #pragma unroll
    for (int i = 0; i < 8; ++i) { a[i] = a0[i]; a[i + 8] = a0[8 + i]; }

    #pragma unroll
    for (int o = 0; o < 3; ++o) {
      #pragma unroll
      for (int t = 0; t < 2; ++t) {
        int col = wave * 32 + t * 16 + l15;
        const __bf16* bp = wt + ((size_t)(o * HID + col)) * KPAD + kc * 32 + half * 16;
        bf16x16 bm;
        #pragma unroll
        for (int i = 0; i < 16; ++i) bm[i] = bp[i];
        acc[o * 2 + t] = __builtin_amdgcn_wmma_f32_16x16x32_bf16(
            false, a, false, bm, (short)0, acc[o * 2 + t], false, false);
      }
    }
  }

  const float* biases[3] = {bq, bk, bv};
  #pragma unroll
  for (int o = 0; o < 3; ++o) {
    #pragma unroll
    for (int t = 0; t < 2; ++t) {
      int col = wave * 32 + t * 16 + l15;
      float bb = biases[o][col];
      #pragma unroll
      for (int j = 0; j < 8; ++j) {
        int grow = row0 + j + half * 8;
        float v  = acc[o * 2 + t][j] + bb;
        if (o == 0)      qout[((size_t)b * SEQ + grow) * HID + col]  = (__bf16)v;
        else if (o == 1) kout[((size_t)b * SEQ + grow) * HID + col]  = (__bf16)v;
        else             vtout[((size_t)b * HID + col) * SEQ + grow] = (__bf16)v;
      }
    }
  }
}

// ------------------------- kernel 2: flash attention -----------------------
// grid = B*(L/64), 128 threads (4 waves); each wave owns 16 q-rows x 256 cols.
// Double-buffered async K/V staging: chunk kc+1 streams into LDS behind the
// 32 WMMAs + softmax of chunk kc; s_wait_asynccnt 16 retires only the
// previous chunk's loads (async loads complete in order).

// Per-thread async transfers per chunk: 8 (K) + 8 (V) = 16.
__device__ __forceinline__ void stage_chunk(int tid, const __bf16* kbase,
                                            const __bf16* vbase, int kc,
                                            unsigned sK_lds, unsigned sV_lds) {
  const char* ksrc = (const char*)(kbase + (size_t)kc * 32 * HID);  // 16 KB contig
  #pragma unroll
  for (int i = 0; i < 8; ++i) {
    int idx = tid + i * 128;                       // 0..1023
    async_load_b128(sK_lds + idx * 16, ksrc + (size_t)idx * 16);
  }
  #pragma unroll
  for (int i = 0; i < 8; ++i) {
    int lin = tid + i * 128;                       // 0..1023
    int row = lin >> 2, seg = lin & 3;             // 256 rows x 4 b128 segs
    const char* vsrc = (const char*)(vbase + (size_t)row * SEQ + kc * 32) + seg * 16;
    async_load_b128(sV_lds + row * 64 + seg * 16, vsrc);
  }
}

__global__ void __launch_bounds__(128)
attn_kernel(const __bf16* __restrict__ q, const __bf16* __restrict__ k,
            const __bf16* __restrict__ vt, const float* __restrict__ mask,
            float* __restrict__ out) {
  int blk   = blockIdx.x;          // 0..255
  int b     = blk >> 6;
  int qb    = blk & 63;
  int tid   = threadIdx.x;
  int wave  = tid >> 5, lane = tid & 31;
  int half  = lane >> 4;
  int l15   = lane & 15;
  int qrow0 = qb * 64 + wave * 16;

  __shared__ __align__(16) __bf16 sK[2][32 * HID];   // 2 x 16 KB key chunks
  __shared__ __align__(16) __bf16 sV[2][HID * 32];   // 2 x 16 KB V^T chunks
  __shared__ __align__(16) __bf16 sP[4][16 * 32];    // per-wave P transpose

  const __bf16* qbase = q  + (size_t)b * SEQ * HID;
  const __bf16* kbase = k  + (size_t)b * SEQ * HID;
  const __bf16* vbase = vt + (size_t)b * HID * SEQ;
  const float*  mbase = mask + (size_t)b * SEQ * SEQ;

  // Hold this wave's q tile in registers as 8 A-operand chunks (K=256).
  bf16x16 qa[8];
  {
    const __bf16* qrow = qbase + (size_t)(qrow0 + l15) * HID;
    #pragma unroll
    for (int hc = 0; hc < 8; ++hc) {
      int kb = hc * 32 + half * 16;
      #pragma unroll
      for (int i = 0; i < 8; ++i) { qa[hc][i] = qrow[kb + i]; qa[hc][i + 8] = qrow[kb + 8 + i]; }
    }
  }

  f32x8 acc[16];
  #pragma unroll
  for (int t = 0; t < 16; ++t) acc[t] = f32x8{};
  float mrow[8], lrow[8];
  #pragma unroll
  for (int j = 0; j < 8; ++j) { mrow[j] = -1e30f; lrow[j] = 0.0f; }

  unsigned sK_lds[2] = { (unsigned)(size_t)&sK[0][0], (unsigned)(size_t)&sK[1][0] };
  unsigned sV_lds[2] = { (unsigned)(size_t)&sV[0][0], (unsigned)(size_t)&sV[1][0] };
  int ksel = half * 16;
  constexpr int NCHUNK = SEQ / 32;

  // Prologue: start chunk 0 into buffer 0.
  stage_chunk(tid, kbase, vbase, 0, sK_lds[0], sV_lds[0]);

  for (int kc = 0; kc < NCHUNK; ++kc) {
    int buf = kc & 1;
    // Issue next chunk into the other buffer (its previous consumer finished
    // at the trailing barrier of iteration kc-1), then retire chunk kc.
    if (kc + 1 < NCHUNK) {
      stage_chunk(tid, kbase, vbase, kc + 1, sK_lds[buf ^ 1], sV_lds[buf ^ 1]);
      wait_async16();
    } else {
      wait_async0();
    }
    __syncthreads();

    const __bf16* sKb = &sK[buf][0];
    const __bf16* sVb = &sV[buf][0];

    // ---- S = q @ K^T for 16x32 tile (two 16x16 WMMA tiles, K=256)
    f32x8 s0{}, s1{};
    #pragma unroll
    for (int hc = 0; hc < 8; ++hc) {
      int hk = hc * 32 + ksel;
      const __bf16* p0 = &sKb[l15 * HID + hk];
      const __bf16* p1 = &sKb[(l15 + 16) * HID + hk];
      bf16x16 b0, b1;
      #pragma unroll
      for (int i = 0; i < 16; ++i) { b0[i] = p0[i]; b1[i] = p1[i]; }
      s0 = __builtin_amdgcn_wmma_f32_16x16x32_bf16(false, qa[hc], false, b0, (short)0, s0, false, false);
      s1 = __builtin_amdgcn_wmma_f32_16x16x32_bf16(false, qa[hc], false, b1, (short)0, s1, false, false);
    }

    // ---- mask + online softmax (rows j+8*half live in this lane's VGPR j)
    float p0v[8], p1v[8], corr[8];
    int c0 = kc * 32 + l15;
    #pragma unroll
    for (int j = 0; j < 8; ++j) {
      int grow = qrow0 + j + half * 8;
      const float* mrp = mbase + (size_t)grow * SEQ;
      float sv0 = s0[j] + mrp[c0]      * -100000.0f;
      float sv1 = s1[j] + mrp[c0 + 16] * -100000.0f;
      if (kc + 1 < NCHUNK) {           // prefetch next chunk's mask segment
        __builtin_prefetch(mrp + c0 + 32, 0, 0);
      }
      float mj = fmaxf(sv0, sv1);
      mj = fmaxf(mj, __shfl_xor(mj, 1));
      mj = fmaxf(mj, __shfl_xor(mj, 2));
      mj = fmaxf(mj, __shfl_xor(mj, 4));
      mj = fmaxf(mj, __shfl_xor(mj, 8));
      float mn = fmaxf(mrow[j], mj);
      float cj = __expf(mrow[j] - mn);
      float e0 = __expf(sv0 - mn);
      float e1 = __expf(sv1 - mn);
      float rs = e0 + e1;
      rs += __shfl_xor(rs, 1);
      rs += __shfl_xor(rs, 2);
      rs += __shfl_xor(rs, 4);
      rs += __shfl_xor(rs, 8);
      lrow[j] = lrow[j] * cj + rs;
      mrow[j] = mn;
      corr[j] = cj; p0v[j] = e0; p1v[j] = e1;
    }
    #pragma unroll
    for (int t = 0; t < 16; ++t)
      #pragma unroll
      for (int j = 0; j < 8; ++j) acc[t][j] *= corr[j];

    // ---- transpose P (C-layout -> A-layout) through per-wave LDS scratch
    __bf16* myP = &sP[wave][0];
    #pragma unroll
    for (int j = 0; j < 8; ++j) {
      int prow = j + half * 8;
      myP[prow * 32 + l15]      = (__bf16)p0v[j];
      myP[prow * 32 + 16 + l15] = (__bf16)p1v[j];
    }
    bf16x16 pa;
    {
      const __bf16* pr = &myP[l15 * 32 + half * 16];
      #pragma unroll
      for (int i = 0; i < 8; ++i) { pa[i] = pr[i]; pa[i + 8] = pr[8 + i]; }
    }

    // ---- out += P @ V  (16 N-tiles, K=32)
    #pragma unroll
    for (int t = 0; t < 16; ++t) {
      const __bf16* vp = &sVb[(t * 16 + l15) * 32 + ksel];
      bf16x16 vb;
      #pragma unroll
      for (int i = 0; i < 16; ++i) vb[i] = vp[i];
      acc[t] = __builtin_amdgcn_wmma_f32_16x16x32_bf16(false, pa, false, vb, (short)0, acc[t], false, false);
    }
    __syncthreads();   // all waves done with buf before it is restaged
  }

  // ---- epilogue: normalize and store fp32
  #pragma unroll
  for (int j = 0; j < 8; ++j) {
    float rl = 1.0f / lrow[j];
    int grow = qrow0 + j + half * 8;
    float* orow = out + ((size_t)b * SEQ + grow) * HID;
    #pragma unroll
    for (int t = 0; t < 16; ++t) orow[t * 16 + l15] = acc[t][j] * rl;
  }
}

// --------------------------------- launch ----------------------------------
extern "C" void kernel_launch(void* const* d_in, const int* in_sizes, int n_in,
                              void* d_out, int out_size, void* d_ws, size_t ws_size,
                              hipStream_t stream) {
  const float* emb  = (const float*)d_in[0];
  const float* mask = (const float*)d_in[1];
  const float* Wq   = (const float*)d_in[2];
  const float* bq   = (const float*)d_in[3];
  const float* Wk   = (const float*)d_in[4];
  const float* bk   = (const float*)d_in[5];
  const float* Wv   = (const float*)d_in[6];
  const float* bv   = (const float*)d_in[7];
  float* out = (float*)d_out;

  char* ws = (char*)d_ws;
  __bf16* wt   = (__bf16*)(ws + WT_OFF);
  __bf16* qbuf = (__bf16*)(ws + Q_OFF);
  __bf16* kbuf = (__bf16*)(ws + K_OFF);
  __bf16* vbuf = (__bf16*)(ws + V_OFF);

  wt_kernel<<<3 * HID, KPAD, 0, stream>>>(Wq, Wk, Wv, wt);
  qkv_proj<<<BATCH * (SEQ / 16), 256, 0, stream>>>(emb, wt, bq, bk, bv,
                                                   qbuf, kbuf, vbuf);
  attn_kernel<<<BATCH * (SEQ / 64), 128, 0, stream>>>(qbuf, kbuf, vbuf, mask, out);
}
